// HGNNLayer_13735305413432
// MI455X (gfx1250) — matrix operand order repaired and checked by reading, not compiled, not run on hardware
//
#include <hip/hip_runtime.h>

// ---------------------------------------------------------------------------
// HGNN layer: out = (Dv^-1/2 H De^-1 H^T Dv^-1/2 X) W^T + b
// N=8192 nodes, E=1024 hyperedges, F=256 features.
// GEMMs: v_wmma_f32_16x16x32_bf16 (wave32), tiles DMA'd into LDS with
// global_load_async_to_lds_b128 (ASYNCcnt) + double buffering -> one barrier
// per K-chunk. H is binary -> exact in bf16. B operands stored column-major
// (Bt row-major) so all LDS traffic is b128-granular.
// ---------------------------------------------------------------------------

#define N_NODES 8192
#define E_EDGES 1024
#define FDIM    256

typedef __bf16 bf16_t;
typedef __attribute__((ext_vector_type(16))) __bf16 v16bf;
typedef __attribute__((ext_vector_type(8)))  float  v8f;

static __device__ __forceinline__ bf16_t f32_to_bf16(float f) {
    unsigned u = __float_as_uint(f);
    u += 0x7FFFu + ((u >> 16) & 1u);          // round to nearest even
    unsigned short s = (unsigned short)(u >> 16);
    return __builtin_bit_cast(bf16_t, s);
}

// Async DMA: 16B global -> LDS, per lane. Tracked by ASYNCcnt.
static __device__ __forceinline__ void async_copy_b128(unsigned lds_off,
                                                       const void* gptr) {
    asm volatile("global_load_async_to_lds_b128 %0, %1, off"
                 :: "v"(lds_off), "v"(gptr)
                 : "memory");
}
static __device__ __forceinline__ void wait_async0() {
    asm volatile("s_wait_asynccnt 0x0" ::: "memory");
}

// ---------------------------------------------------------------------------
// dv_inv_sqrt[i] = 1/sqrt(rowsum(H[i,:])); one wave32 per row.
// ---------------------------------------------------------------------------
__global__ __launch_bounds__(256) void dv_kernel(const float* __restrict__ H,
                                                 float* __restrict__ dv_is) {
    int wave = (blockIdx.x * blockDim.x + threadIdx.x) >> 5;
    int lane = threadIdx.x & 31;
    if (wave >= N_NODES) return;
    const float* row = H + (size_t)wave * E_EDGES;
    float s = 0.f;
    for (int e = lane; e < E_EDGES; e += 32) s += row[e];
    #pragma unroll
    for (int off = 16; off > 0; off >>= 1) s += __shfl_down(s, off, 32);
    if (lane == 0) dv_is[wave] = rsqrtf(s);
}

// ---------------------------------------------------------------------------
// de: deterministic two-phase column sums (no float atomics).
// ---------------------------------------------------------------------------
__global__ __launch_bounds__(256) void de_partial_kernel(const float* __restrict__ H,
                                                         float* __restrict__ part) {
    int j  = blockIdx.x * 256 + threadIdx.x;   // column 0..1023
    int rc = blockIdx.y;                       // row chunk 0..31 (256 rows each)
    const float* base = H + (size_t)rc * 256 * E_EDGES + j;
    float s = 0.f;
    for (int i = 0; i < 256; ++i) s += base[(size_t)i * E_EDGES];
    part[rc * E_EDGES + j] = s;
}

__global__ __launch_bounds__(256) void de_final_kernel(const float* __restrict__ part,
                                                       float* __restrict__ de_inv) {
    int j = blockIdx.x * 256 + threadIdx.x;
    float s = 0.f;
    #pragma unroll
    for (int c = 0; c < 32; ++c) s += part[c * E_EDGES + j];
    de_inv[j] = 1.0f / s;
}

// ---------------------------------------------------------------------------
// XsT[f][i] = bf16(X[i][f] * dv_inv_sqrt[i])   (transpose-convert + scale)
// ---------------------------------------------------------------------------
__global__ void xst_kernel(const float* __restrict__ X,
                           const float* __restrict__ dv_is,
                           bf16_t* __restrict__ XsT) {
    __shared__ float tile[32][33];
    int i0 = blockIdx.x * 32;                  // node dim
    int f0 = blockIdx.y * 32;                  // feature dim
    int tx = threadIdx.x, ty = threadIdx.y;    // (32, 8)
    #pragma unroll
    for (int r = ty; r < 32; r += 8)
        tile[r][tx] = X[(size_t)(i0 + r) * FDIM + f0 + tx];
    __syncthreads();
    float s = dv_is[i0 + tx];
    #pragma unroll
    for (int r = ty; r < 32; r += 8)
        XsT[(size_t)(f0 + r) * N_NODES + i0 + tx] = f32_to_bf16(tile[tx][r] * s);
}

// ---------------------------------------------------------------------------
// Plain fp32 -> bf16 convert (H and W); 4 elems/thread.
// ---------------------------------------------------------------------------
__global__ __launch_bounds__(256) void conv_kernel(const float* __restrict__ S,
                                                   bf16_t* __restrict__ D) {
    size_t idx = ((size_t)blockIdx.x * blockDim.x + threadIdx.x) * 4;
    float4 v = *(const float4*)(S + idx);
    union { bf16_t h[4]; uint2 u; } o;
    o.h[0] = f32_to_bf16(v.x);
    o.h[1] = f32_to_bf16(v.y);
    o.h[2] = f32_to_bf16(v.z);
    o.h[3] = f32_to_bf16(v.w);
    *(uint2*)(D + idx) = o.u;
}

// ---------------------------------------------------------------------------
// Ht[e][i] = bf16(H[i][e]); tiled 32x32 LDS transpose, coalesced both ways.
// ---------------------------------------------------------------------------
__global__ void htrans_kernel(const float* __restrict__ H, bf16_t* __restrict__ Ht) {
    __shared__ float tile[32][33];
    int i0 = blockIdx.x * 32;                  // node dim
    int e0 = blockIdx.y * 32;                  // edge dim
    int tx = threadIdx.x, ty = threadIdx.y;    // (32, 8)
    #pragma unroll
    for (int r = ty; r < 32; r += 8)
        tile[r][tx] = H[(size_t)(i0 + r) * E_EDGES + e0 + tx];
    __syncthreads();
    #pragma unroll
    for (int r = ty; r < 32; r += 8)
        Ht[(size_t)(e0 + r) * N_NODES + i0 + tx] = f32_to_bf16(tile[tx][r]);
}

// ---------------------------------------------------------------------------
// bf16 transpose: D[c][r] = S[r][c]   (for M -> Mt, 1024x256 -> 256x1024)
// ---------------------------------------------------------------------------
__global__ void bt16_kernel(const bf16_t* __restrict__ S, bf16_t* __restrict__ D,
                            int rows, int cols) {
    __shared__ bf16_t tile[32][33];
    int r0 = blockIdx.x * 32;
    int c0 = blockIdx.y * 32;
    int tx = threadIdx.x, ty = threadIdx.y;
    #pragma unroll
    for (int r = ty; r < 32; r += 8)
        tile[r][tx] = S[(size_t)(r0 + r) * cols + c0 + tx];
    __syncthreads();
    #pragma unroll
    for (int r = ty; r < 32; r += 8)
        D[(size_t)(c0 + r) * rows + r0 + tx] = tile[tx][r];
}

// ---------------------------------------------------------------------------
// bf16 WMMA GEMM:  C[M x Nn] = A[M x K] * Bt[Nn x K]^T    (both row-major bf16)
// epilogue: *rowScale[m], +bias[n], store fp32 and/or bf16.
// Block = 256 threads (8 waves), block tile 128x64, K chunk 32, double-buffered
// LDS filled by global_load_async_to_lds_b128; one barrier per chunk.
// CDNA5 wave32 fragment layouts per ISA 7.12.2:
//   A 16x32 bf16: lane L<16 -> row L, K in {hi*8+0..7, hi*8+16..23} (hi=lane>>4)
//   B 32x16 bf16: lane L    -> col L&15, K in {hi*8+0..7, hi*8+16..23}  (Bs holds Bt rows)
//   C 16x16 f32 : vgpr v    -> row v + 8*hi, col lane&15
// ---------------------------------------------------------------------------
#define BM 128
#define BN 64
#define BK 32
#define LDA 40   /* bf16 elems; 80B rows keep 16B alignment, dodge bank conflicts */
#define LDB 40

__global__ __launch_bounds__(256)
void gemm_bf16_wmma(const bf16_t* __restrict__ A, const bf16_t* __restrict__ Bt,
                    int M, int Nn, int K,
                    const float* __restrict__ rowScale,
                    const float* __restrict__ bias,
                    float* __restrict__ Cf, bf16_t* __restrict__ Cb) {
    __shared__ bf16_t As[2][BM * LDA];
    __shared__ bf16_t Bs[2][BN * LDB];   // Bs[p][n][k]

    const int tid    = threadIdx.x;
    const int lane   = tid & 31;
    const int waveId = tid >> 5;
    const int wm     = waveId >> 1;     // 0..3 -> 32-row band
    const int wn     = waveId & 1;      // 0..1 -> 32-col band
    const int r16    = lane & 15;
    const int hi     = lane >> 4;

    const int rowBase = blockIdx.x * BM;
    const int colBase = blockIdx.y * BN;

    // cooperative DMA assignments (16B per async op)
    const int arow  = tid >> 1;             // 0..127
    const int ahalf = (tid & 1) * 16;       // 0 or 16 bf16
    const int brow  = tid >> 2;             // 0..63 (n)
    const int bkp   = (tid & 3) * 8;        // 0,8,16,24 (k)

    const bf16_t* aSrc = A  + (size_t)(rowBase + arow) * K + ahalf;
    const bf16_t* bSrc = Bt + (size_t)(colBase + brow) * K + bkp;

    // LDS byte addresses for the async DMA destination (flat addr[31:0] is the
    // wave-relative LDS address per ISA 10.2 aperture mapping)
    unsigned ldsA[2], ldsB[2];
    #pragma unroll
    for (int p = 0; p < 2; ++p) {
        ldsA[p] = (unsigned)(uintptr_t)&As[p][arow * LDA + ahalf];
        ldsB[p] = (unsigned)(uintptr_t)&Bs[p][brow * LDB + bkp];
    }

    v8f acc[2][2] = {};

    const int nch = K / BK;
    // prologue: chunk 0 -> buffer 0
    async_copy_b128(ldsA[0],      aSrc);
    async_copy_b128(ldsA[0] + 16, aSrc + 8);
    async_copy_b128(ldsB[0],      bSrc);

    for (int i = 0; i < nch; ++i) {
        const int p = i & 1;
        wait_async0();          // this wave's DMAs for chunk i (and earlier) landed
        __syncthreads();        // ... and every other wave's too
        if (i + 1 < nch) {      // stream chunk i+1 into the other buffer while
            const int np = p ^ 1;     // WMMA consumes chunk i (safe: barrier above
            const int k1 = (i + 1) * BK;  // proves buffer np's readers finished)
            async_copy_b128(ldsA[np],      aSrc + k1);
            async_copy_b128(ldsA[np] + 16, aSrc + k1 + 8);
            async_copy_b128(ldsB[np],      bSrc + k1);
        }

        v16bf afrag[2], bfrag[2];
        #pragma unroll
        for (int mt = 0; mt < 2; ++mt) {   // two contiguous ds_load_b128 per frag
            const bf16_t* ap = &As[p][(wm * 32 + mt * 16 + r16) * LDA + hi * 8];
            uint4* dst = (uint4*)&afrag[mt];
            dst[0] = *(const uint4*)(ap);
            dst[1] = *(const uint4*)(ap + 16);
        }
        #pragma unroll
        for (int nt = 0; nt < 2; ++nt) {
            const bf16_t* bp = &Bs[p][(wn * 32 + nt * 16 + r16) * LDB + hi * 8];
            uint4* dst = (uint4*)&bfrag[nt];
            dst[0] = *(const uint4*)(bp);
            dst[1] = *(const uint4*)(bp + 16);
        }
        #pragma unroll
        for (int mt = 0; mt < 2; ++mt)
            #pragma unroll
            for (int nt = 0; nt < 2; ++nt)
                acc[mt][nt] = __builtin_amdgcn_wmma_f32_16x16x32_bf16(
                    false, afrag[mt], false, bfrag[nt],
                    (short)0, acc[mt][nt], false, false);
    }

    // fused epilogue: diagonal row scaling + bias + store (f32 and/or bf16)
    #pragma unroll
    for (int mt = 0; mt < 2; ++mt) {
        #pragma unroll
        for (int v = 0; v < 8; ++v) {
            int gm = rowBase + wm * 32 + mt * 16 + hi * 8 + v;
            float s = rowScale ? rowScale[gm] : 1.0f;
            #pragma unroll
            for (int nt = 0; nt < 2; ++nt) {
                int gn = colBase + wn * 32 + nt * 16 + r16;
                float val = acc[mt][nt][v] * s;
                if (bias) val += bias[gn];
                size_t o = (size_t)gm * Nn + gn;
                if (Cf) Cf[o] = val;
                if (Cb) Cb[o] = f32_to_bf16(val);
            }
        }
    }
}

// ---------------------------------------------------------------------------
extern "C" void kernel_launch(void* const* d_in, const int* in_sizes, int n_in,
                              void* d_out, int out_size, void* d_ws, size_t ws_size,
                              hipStream_t stream) {
    (void)in_sizes; (void)n_in; (void)out_size; (void)ws_size;
    const float* X = (const float*)d_in[0];   // [8192, 256]
    const float* H = (const float*)d_in[1];   // [8192, 1024]
    const float* W = (const float*)d_in[2];   // [256, 256] (F_OUT x F_IN)
    const float* b = (const float*)d_in[3];   // [256]
    float* out = (float*)d_out;               // [8192, 256] fp32

    // carve workspace (all regions fully written before read each call)
    size_t off = 0;
    char* base = (char*)d_ws;
    auto take = [&](size_t bytes) -> char* {
        char* p = base + off;
        off += (bytes + 255) & ~(size_t)255;
        return p;
    };
    float*  dv_is   = (float*)take((size_t)N_NODES * 4);
    float*  de_inv  = (float*)take((size_t)E_EDGES * 4);
    float*  de_part = (float*)take((size_t)32 * E_EDGES * 4);
    bf16_t* XsT     = (bf16_t*)take((size_t)FDIM * N_NODES * 2);    // [256][8192]
    bf16_t* Ht      = (bf16_t*)take((size_t)E_EDGES * N_NODES * 2); // [1024][8192]
    bf16_t* Hb      = (bf16_t*)take((size_t)N_NODES * E_EDGES * 2); // [8192][1024]
    bf16_t* Mb      = (bf16_t*)take((size_t)E_EDGES * FDIM * 2);    // [1024][256]
    bf16_t* MbT     = (bf16_t*)take((size_t)FDIM * E_EDGES * 2);    // [256][1024]
    bf16_t* Xn      = (bf16_t*)take((size_t)N_NODES * FDIM * 2);    // [8192][256]
    bf16_t* Wb      = (bf16_t*)take((size_t)FDIM * FDIM * 2);       // [256][256] = (W^T)^T

    // degrees
    dv_kernel<<<N_NODES * 32 / 256, 256, 0, stream>>>(H, dv_is);
    de_partial_kernel<<<dim3(E_EDGES / 256, 32), 256, 0, stream>>>(H, de_part);
    de_final_kernel<<<E_EDGES / 256, 256, 0, stream>>>(de_part, de_inv);

    // precision/layout conversions
    xst_kernel<<<dim3(N_NODES / 32, FDIM / 32), dim3(32, 8), 0, stream>>>(X, dv_is, XsT);
    conv_kernel<<<(size_t)N_NODES * E_EDGES / 4 / 256, 256, 0, stream>>>(H, Hb);
    conv_kernel<<<(size_t)FDIM * FDIM / 4 / 256, 256, 0, stream>>>(W, Wb);
    htrans_kernel<<<dim3(N_NODES / 32, E_EDGES / 32), dim3(32, 8), 0, stream>>>(H, Ht);

    // GEMM 1: M_e = De^-1 * (H^T @ Xs)        [1024 x 256], K = 8192
    gemm_bf16_wmma<<<dim3(E_EDGES / BM, FDIM / BN), 256, 0, stream>>>(
        Ht, XsT, E_EDGES, FDIM, N_NODES, de_inv, nullptr, nullptr, Mb);

    // Mt for GEMM2's column-major B operand   [256 x 1024]
    bt16_kernel<<<dim3(E_EDGES / 32, FDIM / 32), dim3(32, 8), 0, stream>>>(
        Mb, MbT, E_EDGES, FDIM);

    // GEMM 2: Xn = Dv^-1/2 * (H @ M_e)        [8192 x 256], K = 1024
    gemm_bf16_wmma<<<dim3(N_NODES / BM, FDIM / BN), 256, 0, stream>>>(
        Hb, MbT, N_NODES, FDIM, E_EDGES, dv_is, nullptr, nullptr, Xn);

    // GEMM 3: out = Xn @ W^T + b              [8192 x 256], K = 256, fp32 out
    gemm_bf16_wmma<<<dim3(N_NODES / BM, FDIM / BN), 256, 0, stream>>>(
        Xn, Wb, N_NODES, FDIM, FDIM, nullptr, b, out, nullptr);
}
